// Real_Local_Encoder_14422500180561
// MI455X (gfx1250) — matrix operand order repaired and checked by reading, not compiled, not run on hardware
//
#include <hip/hip_runtime.h>
#include <hip/hip_bf16.h>
#include <math.h>

// ---------------------------------------------------------------------------
// Sizes from the reference
// ---------------------------------------------------------------------------
#define N_SEQ   8192
#define F_IN    128
#define D_MODEL 256
#define D_INNER 512
#define D_STATE 16
#define D_CONV  4
#define DT_RANK 16
#define NCHUNK  32
#define CHUNK   (N_SEQ / NCHUNK)   // 256

typedef __bf16 bf16;
typedef __attribute__((ext_vector_type(16))) __bf16 v16bf;
typedef __attribute__((ext_vector_type(8)))  float  v8f;
typedef __attribute__((ext_vector_type(4)))  unsigned int v4u;
typedef __attribute__((ext_vector_type(8)))  int v8i;
typedef __attribute__((ext_vector_type(4)))  int v4i;

union FragB { uint4 q[2]; v16bf v; };

__device__ __forceinline__ float sigmf(float x) { return 1.0f / (1.0f + __expf(-x)); }

#if __has_builtin(__builtin_amdgcn_tensor_load_to_lds) && __has_builtin(__builtin_amdgcn_s_wait_tensorcnt)
#define USE_TDM 1
#else
#define USE_TDM 0
#endif

// ---------------------------------------------------------------------------
// Generic 128x128-tile WMMA GEMM:  C[M x N] = A[M x K] * B[N x K]^T
// B staged into LDS by the Tensor Data Mover (double-buffered), with TDM
// padding (64B row + 16B pad = 80B stride) to spread LDS banks.
// Block = 8 waves; each wave produces a 16(M) x 128(N) strip (8 accumulators).
// ---------------------------------------------------------------------------
#define BS_STRIDE 40   // 32 bf16 + 8 pad (80B row), matches TDM pad settings

#if USE_TDM
// D# for a 2-D tile: rows 0..127 (dim1), k window of 32 (dim0), 2-byte elems.
// g1 word0: data_size=1(2B)<<16 | pad_enable(1<<20) | pad_interval=3 (16 DW
// = 64B, one tile row) <<22 | pad_amount=3 (4 DW = 16B) <<25.
__device__ __forceinline__ void tdm_load_tile(unsigned int lds_off, unsigned long long gaddr, int K)
{
    v4u g0 = { 1u,                                  // count=1, user mode
               lds_off,                             // LDS dest byte offset
               (unsigned int)gaddr,                 // global_addr[31:0]
               (unsigned int)((gaddr >> 32) & 0x01FFFFFFu) | (2u << 30) }; // addr[56:32] | type=2
    v8i g1 = { (int)((1u << 16) | (1u << 20) | (3u << 22) | (3u << 25)),
               (int)((unsigned)(K & 0xFFFF) << 16),           // tensor_dim0[15:0] @ 63:48
               (int)(((unsigned)K >> 16) | (128u << 16)),     // dim0[31:16] @ 79:64 | dim1[15:0] @ 95:80
               (int)(32u << 16),                              // dim1 hi=0 | tile_dim0=32 @ 127:112
               (int)128,                                      // tile_dim1=128 | tile_dim2=0
               (int)K,                                        // tensor_dim0_stride[31:0]
               0, 0 };
    v4i z4 = { 0, 0, 0, 0 };
    v8i z8 = { 0, 0, 0, 0, 0, 0, 0, 0 };
    __builtin_amdgcn_tensor_load_to_lds(g0, g1, z4, z4, z8, 0);
}
#endif

template <bool AF32, class Epi>
__global__ __launch_bounds__(256) void gemm_wmma(const void* __restrict__ Aun,
                                                 const bf16* __restrict__ Bmat,
                                                 int K, Epi epi)
{
    __shared__ bf16 Bs[2][128 * BS_STRIDE];

    const int tid  = threadIdx.x;
    const int w    = tid >> 5;
    const int lane = tid & 31;
    const int l16  = lane & 15;
    const int lh   = lane >> 4;
    const int m0   = blockIdx.x * 128;
    const int n0   = blockIdx.y * 128;
    const int mrow = m0 + w * 16 + l16;

    v8f acc[8];
    const v8f vzero = {0.f, 0.f, 0.f, 0.f, 0.f, 0.f, 0.f, 0.f};
#pragma unroll
    for (int j = 0; j < 8; ++j) acc[j] = vzero;

    const float* Af = (const float*)Aun;
    const bf16*  Ab = (const bf16*)Aun;
    const int nsteps = K / 32;

#if USE_TDM
    const unsigned int ldsBase0 = (unsigned int)(size_t)(void*)&Bs[0][0];
    const unsigned int ldsBase1 = (unsigned int)(size_t)(void*)&Bs[1][0];
    const unsigned long long gbase =
        (unsigned long long)(size_t)Bmat + (unsigned long long)(size_t)n0 * K * 2ull;

    if (w == 0) {
        tdm_load_tile(ldsBase0, gbase, K);
        __builtin_amdgcn_s_wait_tensorcnt((short)0);
    }
    __syncthreads();
#endif

    for (int step = 0; step < nsteps; ++step) {
        const int kb  = step * 32;
        const int cur = step & 1;

#if USE_TDM
        // kick off DMA for the next tile into the other buffer
        if (w == 0 && step + 1 < nsteps)
            tdm_load_tile(cur ? ldsBase0 : ldsBase1,
                          gbase + (unsigned long long)(kb + 32) * 2ull, K);
#else
        {
            const int col  = tid >> 1;
            const int half = tid & 1;
            const uint4* s4 = (const uint4*)(Bmat + (size_t)(n0 + col) * K + kb + 16 * half);
            uint4* d4 = (uint4*)(&Bs[0][0] + col * BS_STRIDE + 16 * half);
            d4[0] = s4[0];
            d4[1] = s4[1];
        }
        __syncthreads();
#endif

        // ---- A fragment: lane holds row `mrow`, K = {kA..kA+7, kA+16..kA+23}
        FragB a;
        if (AF32) {
            const float* ap = Af + (size_t)mrow * K + kb + 8 * lh;
            float4 x0 = *(const float4*)(ap + 0);
            float4 x1 = *(const float4*)(ap + 4);
            float4 x2 = *(const float4*)(ap + 16);
            float4 x3 = *(const float4*)(ap + 20);
            a.v[0]  = (bf16)x0.x; a.v[1]  = (bf16)x0.y; a.v[2]  = (bf16)x0.z; a.v[3]  = (bf16)x0.w;
            a.v[4]  = (bf16)x1.x; a.v[5]  = (bf16)x1.y; a.v[6]  = (bf16)x1.z; a.v[7]  = (bf16)x1.w;
            a.v[8]  = (bf16)x2.x; a.v[9]  = (bf16)x2.y; a.v[10] = (bf16)x2.z; a.v[11] = (bf16)x2.w;
            a.v[12] = (bf16)x3.x; a.v[13] = (bf16)x3.y; a.v[14] = (bf16)x3.z; a.v[15] = (bf16)x3.w;
            if (kb + 32 < K)  // streaming A rows -> global_prefetch_b8
                __builtin_prefetch((const char*)(ap + 32), 0, 1);
        } else {
            const bf16* ap = Ab + (size_t)mrow * K + kb + 8 * lh;
            a.q[0] = *(const uint4*)(ap + 0);
            a.q[1] = *(const uint4*)(ap + 16);
        }

        // ---- preload all 8 B fragments, then run the WMMA chain, so the
        //      scheduler can overlap ds_loads with matrix ops
        const bf16* bbuf = USE_TDM ? &Bs[cur][0] : &Bs[0][0];
        FragB bfr[8];
#pragma unroll
        for (int j = 0; j < 8; ++j) {
            const bf16* bp = bbuf + (j * 16 + l16) * BS_STRIDE + 8 * lh;
            bfr[j].q[0] = *(const uint4*)(bp + 0);
            bfr[j].q[1] = *(const uint4*)(bp + 16);
        }
#pragma unroll
        for (int j = 0; j < 8; ++j)
            acc[j] = __builtin_amdgcn_wmma_f32_16x16x32_bf16(
                false, a.v, false, bfr[j].v, (short)0, acc[j], false, false);

#if USE_TDM
        if (w == 0) __builtin_amdgcn_s_wait_tensorcnt((short)0);
#endif
        __syncthreads();
    }

    epi(acc, m0 + w * 16, n0, lane);
}

// ---- GEMM-1 epilogue: +bias, ReLU, store h, fused BN sum/sumsq atomics ----
struct EpiGcn {
    const float* bias;
    float* h;
    float* sum;
    float* sumsq;
    __device__ void operator()(v8f* acc, int rowBase, int nBase, int lane) const {
        const int l16 = lane & 15, lh = lane >> 4;
#pragma unroll
        for (int j = 0; j < 8; ++j) {
            const int col = nBase + j * 16 + l16;
            const float b = bias[col];
            float s = 0.f, sq = 0.f;
#pragma unroll
            for (int v = 0; v < 8; ++v) {
                const int row = rowBase + lh * 8 + v;
                float val = acc[j][v] + b;
                val = val > 0.f ? val : 0.f;
                h[(size_t)row * D_MODEL + col] = val;
                s += val;
                sq += val * val;
            }
            s  += __shfl_xor(s, 16, 32);
            sq += __shfl_xor(sq, 16, 32);
            if (lh == 0) { atomicAdd(&sum[col], s); atomicAdd(&sumsq[col], sq); }
        }
    }
};

// ---- GEMM-2 epilogue: plain f32 store (xz) ----
struct EpiStore {
    float* out;
    int ldo;
    __device__ void operator()(v8f* acc, int rowBase, int nBase, int lane) const {
        const int l16 = lane & 15, lh = lane >> 4;
#pragma unroll
        for (int j = 0; j < 8; ++j) {
            const int col = nBase + j * 16 + l16;
#pragma unroll
            for (int v = 0; v < 8; ++v) {
                const int row = rowBase + lh * 8 + v;
                out[(size_t)row * ldo + col] = acc[j][v];
            }
        }
    }
};

// ---- GEMM-3 epilogue: + residual h_bn (recomputed from h, scale, shift) ----
struct EpiAddH {
    const float* h;
    const float* scale;
    const float* shift;
    float* out;
    __device__ void operator()(v8f* acc, int rowBase, int nBase, int lane) const {
        const int l16 = lane & 15, lh = lane >> 4;
#pragma unroll
        for (int j = 0; j < 8; ++j) {
            const int col = nBase + j * 16 + l16;
            const float sc = scale[col], sh = shift[col];
#pragma unroll
            for (int v = 0; v < 8; ++v) {
                const int row = rowBase + lh * 8 + v;
                const float hb = h[(size_t)row * D_MODEL + col] * sc + sh;
                out[(size_t)row * D_MODEL + col] = acc[j][v] + hb;
            }
        }
    }
};

// ---------------------------------------------------------------------------
// Small kernels
// ---------------------------------------------------------------------------

// t0T[j, n] = (x @ w_gcn)[n, j], stored transposed in bf16 (so GEMM-1's B
// operand is contiguous in K).
__global__ __launch_bounds__(256) void gcn_xw(const float* __restrict__ x,
                                              const float* __restrict__ wg,
                                              bf16* __restrict__ t0T)
{
    __shared__ float Xs[64 * F_IN];
    const int n0 = blockIdx.x * 64;
    const int t  = threadIdx.x;
    for (int i = t; i < 64 * F_IN; i += 256) Xs[i] = x[(size_t)n0 * F_IN + i];
    __syncthreads();
    const int j = t;  // 0..255
    for (int nl = 0; nl < 64; ++nl) {
        float acc = 0.f;
#pragma unroll 4
        for (int k = 0; k < F_IN; ++k) acc += Xs[nl * F_IN + k] * wg[k * D_MODEL + j];
        t0T[(size_t)j * N_SEQ + n0 + nl] = (bf16)acc;
    }
}

__global__ void cvt_weights(const float* __restrict__ w_in, const float* __restrict__ w_out,
                            bf16* __restrict__ wInB, bf16* __restrict__ wOutB)
{
    const int g = blockIdx.x * 256 + threadIdx.x;           // 0 .. 262143
    wInB[g] = (bf16)w_in[g];
    if (g < D_MODEL * D_INNER) wOutB[g] = (bf16)w_out[g];   // 131072
}

__global__ void bn_finalize(const float* __restrict__ sum, const float* __restrict__ sumsq,
                            const float* __restrict__ gamma, const float* __restrict__ beta,
                            float* __restrict__ scale, float* __restrict__ shift)
{
    const int c = threadIdx.x;
    const float m  = sum[c] * (1.f / N_SEQ);
    const float vr = sumsq[c] * (1.f / N_SEQ) - m * m;
    const float sc = rsqrtf(vr + 1e-5f) * gamma[c];
    scale[c] = sc;
    shift[c] = beta[c] - m * sc;
}

__global__ void bn_apply(const float* __restrict__ h, const float* __restrict__ scale,
                         const float* __restrict__ shift, bf16* __restrict__ hb)
{
    const int g = blockIdx.x * 256 + threadIdx.x;  // 8192*256
    const int c = g & (D_MODEL - 1);
    hb[g] = (bf16)(h[g] * scale[c] + shift[c]);
}

// causal depthwise conv (k=4) + SiLU ; xm = xz[:, :512]
__global__ void conv_silu(const float* __restrict__ xz, const float* __restrict__ cw,
                          const float* __restrict__ cb, float* __restrict__ xc)
{
    const int g = blockIdx.x * 256 + threadIdx.x;  // 8192*512
    const int n = g >> 9;
    const int d = g & (D_INNER - 1);
    float acc = cb[d];
#pragma unroll
    for (int i = 0; i < D_CONV; ++i) {
        const int nn = n - (D_CONV - 1) + i;
        const float xv = (nn >= 0) ? xz[(size_t)nn * (2 * D_INNER) + d] : 0.f;
        acc += xv * cw[d * D_CONV + i];
    }
    xc[g] = acc * sigmf(acc);
}

// per-row: x_dbl = xc_row @ w_xproj^T (48 outs), split into dt_r/B/C,
// then dt = softplus(dt_r @ w_dt^T + b_dt) for all 512 channels.
__global__ __launch_bounds__(512) void xproj_dt(const float* __restrict__ xc,
                                                const float* __restrict__ w_xproj,
                                                const float* __restrict__ w_dt,
                                                const float* __restrict__ b_dt,
                                                float* __restrict__ Bc, float* __restrict__ Cc,
                                                float* __restrict__ dt)
{
    __shared__ float xd[DT_RANK + 2 * D_STATE];  // 48
    const int n = blockIdx.x;
    const int t = threadIdx.x;
    if (t < DT_RANK + 2 * D_STATE) {
        const float* xr = xc + (size_t)n * D_INNER;
        const float* wr = w_xproj + (size_t)t * D_INNER;
        float acc = 0.f;
#pragma unroll 8
        for (int k = 0; k < D_INNER; ++k) acc += xr[k] * wr[k];
        xd[t] = acc;
        if (t >= DT_RANK && t < DT_RANK + D_STATE)       Bc[n * D_STATE + (t - DT_RANK)] = acc;
        else if (t >= DT_RANK + D_STATE)                 Cc[n * D_STATE + (t - DT_RANK - D_STATE)] = acc;
    }
    __syncthreads();
    const int d = t;
    float acc = b_dt[d];
#pragma unroll
    for (int r = 0; r < DT_RANK; ++r) acc += xd[r] * w_dt[d * DT_RANK + r];
    dt[(size_t)n * D_INNER + d] = (acc > 20.f) ? acc : log1pf(__expf(acc));
}

// ---- chunked linear-recurrence scan --------------------------------------
// pass 1: per-(chunk,d,s) partial:  a = exp(A * sum dt),  b = within-chunk h
__global__ void scan_partials(const float* __restrict__ dt, const float* __restrict__ xc,
                              const float* __restrict__ Bc, const float* __restrict__ A_log,
                              float* __restrict__ chunk_a, float* __restrict__ chunk_b)
{
    const int g = blockIdx.x * 256 + threadIdx.x;  // 32 * 8192
    const int chunk = g >> 13;
    const int r = g & 8191;
    const int d = r >> 4, s = r & 15;
    const float A = -__expf(A_log[d * D_STATE + s]);
    float hp = 0.f, sdt = 0.f;
    const int nbase = chunk * CHUNK;
    for (int i = 0; i < CHUNK; ++i) {
        const int n = nbase + i;
        const float dtv = dt[(size_t)n * D_INNER + d];
        const float xcv = xc[(size_t)n * D_INNER + d];
        const float Bv  = Bc[n * D_STATE + s];
        hp = __expf(dtv * A) * hp + dtv * Bv * xcv;
        sdt += dtv;
    }
    chunk_a[g] = __expf(A * sdt);  // product of per-step decays, exactly
    chunk_b[g] = hp;
}

// pass 2: 32-step cross-chunk scan (8192 independent states)
__global__ void scan_starts(const float* __restrict__ ca, const float* __restrict__ cb,
                            float* __restrict__ hstart)
{
    const int idx = blockIdx.x * 256 + threadIdx.x;  // 8192
    float h = 0.f;
    for (int c = 0; c < NCHUNK; ++c) {
        hstart[c * 8192 + idx] = h;
        h = ca[c * 8192 + idx] * h + cb[c * 8192 + idx];
    }
}

// pass 3: re-emit with correct initial state; fuse  y = (ys + D*xc) * silu(z)
// and bf16 conversion for the final WMMA GEMM. Lane layout: 16 lanes/channel
// (one per state); wave32 shfl_xor tree reduces h*C over the 16 states.
__global__ void scan_emit(const float* __restrict__ dt, const float* __restrict__ xc,
                          const float* __restrict__ Bc, const float* __restrict__ Cc,
                          const float* __restrict__ A_log, const float* __restrict__ hstart,
                          const float* __restrict__ Dskip, const float* __restrict__ xz,
                          bf16* __restrict__ yb)
{
    const int g = blockIdx.x * 256 + threadIdx.x;  // 32 * 8192
    const int chunk = g >> 13;
    const int r = g & 8191;
    const int d = r >> 4, s = r & 15;
    const float A = -__expf(A_log[d * D_STATE + s]);
    const float Dd = Dskip[d];
    float h = hstart[g];
    const int nbase = chunk * CHUNK;
    for (int i = 0; i < CHUNK; ++i) {
        const int n = nbase + i;
        const float dtv = dt[(size_t)n * D_INNER + d];
        const float xcv = xc[(size_t)n * D_INNER + d];
        const float Bv  = Bc[n * D_STATE + s];
        const float Cv  = Cc[n * D_STATE + s];
        h = __expf(dtv * A) * h + dtv * Bv * xcv;
        float c = h * Cv;
        c += __shfl_xor(c, 1, 32);
        c += __shfl_xor(c, 2, 32);
        c += __shfl_xor(c, 4, 32);
        c += __shfl_xor(c, 8, 32);
        if (s == 0) {
            const float z = xz[(size_t)n * (2 * D_INNER) + D_INNER + d];
            const float y = (c + Dd * xcv) * (z * sigmf(z));
            yb[(size_t)n * D_INNER + d] = (bf16)y;
        }
    }
}

// ---------------------------------------------------------------------------
// Launch
// ---------------------------------------------------------------------------
extern "C" void kernel_launch(void* const* d_in, const int* in_sizes, int n_in,
                              void* d_out, int out_size, void* d_ws, size_t ws_size,
                              hipStream_t stream)
{
    const float* x        = (const float*)d_in[0];
    const float* adj      = (const float*)d_in[1];
    const float* w_gcn    = (const float*)d_in[2];
    const float* b_gcn    = (const float*)d_in[3];
    const float* bn_gamma = (const float*)d_in[4];
    const float* bn_beta  = (const float*)d_in[5];
    const float* w_in     = (const float*)d_in[6];
    const float* conv_w   = (const float*)d_in[7];
    const float* conv_b   = (const float*)d_in[8];
    const float* w_xproj  = (const float*)d_in[9];
    const float* w_dt     = (const float*)d_in[10];
    const float* b_dt     = (const float*)d_in[11];
    const float* A_log    = (const float*)d_in[12];
    const float* D_skip   = (const float*)d_in[13];
    const float* w_out    = (const float*)d_in[14];
    float* out = (float*)d_out;

    char* base = (char*)d_ws;
    size_t off = 0;
    auto carve = [&](size_t bytes) { char* p = base + off; off += (bytes + 255) & ~(size_t)255; return p; };

    bf16*  t0T     = (bf16*) carve((size_t)D_MODEL * N_SEQ * 2);          // 4 MB
    float* h       = (float*)carve((size_t)N_SEQ * D_MODEL * 4);          // 8 MB
    bf16*  hb      = (bf16*) carve((size_t)N_SEQ * D_MODEL * 2);          // 4 MB
    bf16*  wInB    = (bf16*) carve((size_t)2 * D_INNER * D_MODEL * 2);    // 0.5 MB
    bf16*  wOutB   = (bf16*) carve((size_t)D_MODEL * D_INNER * 2);        // 0.25 MB
    float* stats   = (float*)carve(4 * D_MODEL * 4);                      // sum|sumsq|scale|shift
    float* sum     = stats;
    float* sumsq   = stats + D_MODEL;
    float* scale   = stats + 2 * D_MODEL;
    float* shift   = stats + 3 * D_MODEL;
    float* xz      = (float*)carve((size_t)N_SEQ * 2 * D_INNER * 4);      // 32 MB
    float* xc      = (float*)carve((size_t)N_SEQ * D_INNER * 4);          // 16 MB
    float* Bc      = (float*)carve((size_t)N_SEQ * D_STATE * 4);          // 0.5 MB
    float* Cc      = (float*)carve((size_t)N_SEQ * D_STATE * 4);          // 0.5 MB
    float* dtbuf   = (float*)carve((size_t)N_SEQ * D_INNER * 4);          // 16 MB
    float* chunk_a = (float*)carve((size_t)NCHUNK * D_INNER * D_STATE * 4);
    float* chunk_b = (float*)carve((size_t)NCHUNK * D_INNER * D_STATE * 4);
    float* hstart  = (float*)carve((size_t)NCHUNK * D_INNER * D_STATE * 4);
    bf16*  yb      = (bf16*) carve((size_t)N_SEQ * D_INNER * 2);          // 8 MB

    // BN accumulators must start at zero every call
    (void)hipMemsetAsync(sum, 0, 2 * D_MODEL * sizeof(float), stream);

    // 0) weight conversions + t0T = (x @ w_gcn)^T in bf16
    cvt_weights<<<(2 * D_INNER * D_MODEL) / 256, 256, 0, stream>>>(w_in, w_out, wInB, wOutB);
    gcn_xw<<<N_SEQ / 64, 256, 0, stream>>>(x, w_gcn, t0T);

    // 1) h = relu(adj @ t0 + b_gcn), fused BN partial sums  (WMMA bf16 + TDM)
    gemm_wmma<true, EpiGcn><<<dim3(N_SEQ / 128, D_MODEL / 128), 256, 0, stream>>>(
        (const void*)adj, t0T, N_SEQ, EpiGcn{b_gcn, h, sum, sumsq});

    // 2) BN finalize + apply (bf16 A operand for next GEMM)
    bn_finalize<<<1, D_MODEL, 0, stream>>>(sum, sumsq, bn_gamma, bn_beta, scale, shift);
    bn_apply<<<(N_SEQ * D_MODEL) / 256, 256, 0, stream>>>(h, scale, shift, hb);

    // 3) xz = h_bn @ w_in^T   (WMMA bf16 + TDM)
    gemm_wmma<false, EpiStore><<<dim3(N_SEQ / 128, (2 * D_INNER) / 128), 256, 0, stream>>>(
        (const void*)hb, wInB, D_MODEL, EpiStore{xz, 2 * D_INNER});

    // 4) causal conv + SiLU ; x_proj + dt
    conv_silu<<<(N_SEQ * D_INNER) / 256, 256, 0, stream>>>(xz, conv_w, conv_b, xc);
    xproj_dt<<<N_SEQ, D_INNER, 0, stream>>>(xc, w_xproj, w_dt, b_dt, Bc, Cc, dtbuf);

    // 5) chunked selective scan + gating, emits bf16 y
    scan_partials<<<(NCHUNK * D_INNER * D_STATE) / 256, 256, 0, stream>>>(
        dtbuf, xc, Bc, A_log, chunk_a, chunk_b);
    scan_starts<<<(D_INNER * D_STATE) / 256, 256, 0, stream>>>(chunk_a, chunk_b, hstart);
    scan_emit<<<(NCHUNK * D_INNER * D_STATE) / 256, 256, 0, stream>>>(
        dtbuf, xc, Bc, Cc, A_log, hstart, D_skip, xz, yb);

    // 6) out = y @ w_out^T + h_bn   (WMMA bf16 + TDM)
    gemm_wmma<false, EpiAddH><<<dim3(N_SEQ / 128, D_MODEL / 128), 256, 0, stream>>>(
        (const void*)yb, wOutB, D_INNER, EpiAddH{h, scale, shift, out});

    (void)in_sizes; (void)n_in; (void)out_size; (void)ws_size;
}
